// TransformerVaeEncoder_77884936946297
// MI455X (gfx1250) — matrix operand order repaired and checked by reading, not compiled, not run on hardware
//
#include <hip/hip_runtime.h>
#include <hip/hip_bf16.h>

typedef __attribute__((ext_vector_type(8)))  _Float16 v8h;
typedef __attribute__((ext_vector_type(16))) _Float16 v16h;
typedef __attribute__((ext_vector_type(8)))  float    v8f;

#define NELEM 2097152      // B*S*D = 16*256*512
#define SD    131072       // S*D

__device__ __forceinline__ v8h cvt8(float4 a, float4 b) {
    v8h r;
    r[0] = (_Float16)a.x; r[1] = (_Float16)a.y; r[2] = (_Float16)a.z; r[3] = (_Float16)a.w;
    r[4] = (_Float16)b.x; r[5] = (_Float16)b.y; r[6] = (_Float16)b.z; r[7] = (_Float16)b.w;
    return r;
}

// ACT: 0=none, 1=tanh-gelu, 2=elu+1.  Branch-free, single v_exp_f32 transcendental.
template <int ACT>
__device__ __forceinline__ float act_apply(float x) {
    if (ACT == 1) {
        float u = 0.7978845608028654f * (x + 0.044715f * x * x * x);
        float t = __expf(2.0f * u);
        float th = 1.0f - 2.0f / (t + 1.0f);     // tanh(u), saturates correctly
        return 0.5f * x * (1.0f + th);
    }
    if (ACT == 2) {
        float e = __expf(x);                     // elu(x)+1 for x<=0
        return x > 0.0f ? x + 1.0f : e;
    }
    return x;
}

// ---------------------------------------------------------------- prep: h = x + pos
__global__ __launch_bounds__(256) void prep_kernel(const float* __restrict__ x,
                                                   const float* __restrict__ pos,
                                                   float* __restrict__ h) {
    size_t i = ((size_t)blockIdx.x * 256 + threadIdx.x) * 4;
    float4 a = *(const float4*)(x + i);
    float4 p = *(const float4*)(pos + (i & (SD - 1)));
    float4 o;
    o.x = a.x + p.x; o.y = a.y + p.y; o.z = a.z + p.z; o.w = a.w + p.w;
    *(float4*)(h + i) = o;
}

// ------------------------------------------- generic M=4096 K=512 N=512 WMMA gemm
// C = act(A @ W + bias).  HEADWISE=1 -> W is [H=8][512][64]; 0 -> [512][512].
// Double-buffered LDS, software-pipelined global loads, coalesced b128 epilogue.
template <int HEADWISE, int ACT>
__global__ __launch_bounds__(256) void gemm512_wmma(const float* __restrict__ A,
                                                    const float* __restrict__ W,
                                                    const float* __restrict__ bias,
                                                    float* __restrict__ C) {
    __shared__ __align__(16) char smem[30720];
    _Float16* LA0 = (_Float16*)smem;             // 2 x 128x40 halves (ping-pong)
    _Float16* LB0 = (_Float16*)(smem + 20480);   // 2 x  64x40 halves (ping-pong)
    float*    CT  = (float*)smem;                // epilogue bounce buffer [128][33]

    const int tid  = threadIdx.x;
    const int lane = tid & 31;
    const int wid  = tid >> 5;
    const int blockN = blockIdx.x * 64;
    const int blockM = blockIdx.y * 128;
    const int m16 = lane & 15;
    const int hi  = lane >> 4;

    const int arow = tid >> 1;          // 0..127
    const int aseg = (tid & 1) << 4;    // 0 or 16
    const int brow = tid >> 3;          // 0..31
    const int bcol = (tid & 7) << 3;    // 0..56

    const float* Abase = A + (size_t)(blockM + arow) * 512 + aseg;
    const float* Wbase = HEADWISE
        ? (W + (size_t)blockN * 512 + (size_t)brow * 64 + bcol)
        : (W + (size_t)brow * 512 + blockN + bcol);
    const size_t wstep = HEADWISE ? 64u : 512u;  // per-k-row stride (x32 per tile)

    v8f acc[4] = {};
    float4 a0, a1, a2, a3, g0, g1;

    // prologue: tile k0=0 into registers
    {
        const float4* ap = (const float4*)Abase;
        a0 = ap[0]; a1 = ap[1]; a2 = ap[2]; a3 = ap[3];
        const float4* wp = (const float4*)Wbase;
        g0 = wp[0]; g1 = wp[1];
    }

    for (int ks = 0; ks < 16; ++ks) {
        const int p = ks & 1;
        _Float16* LAp = LA0 + p * 5120;
        _Float16* LBp = LB0 + p * 2560;

        // commit staged registers to LDS buffer p
        *(v8h*)&LAp[arow * 40 + aseg]     = cvt8(a0, a1);
        *(v8h*)&LAp[arow * 40 + aseg + 8] = cvt8(a2, a3);
        v8h bh = cvt8(g0, g1);
#pragma unroll
        for (int j = 0; j < 8; ++j) LBp[(bcol + j) * 40 + brow] = bh[j];

        // issue next tile's global loads (in flight under the WMMAs)
        if (ks < 15) {
            const float4* ap = (const float4*)(Abase + (size_t)(ks + 1) * 32);
            a0 = ap[0]; a1 = ap[1]; a2 = ap[2]; a3 = ap[3];
            const float4* wp = (const float4*)(Wbase + (size_t)(ks + 1) * 32 * wstep);
            g0 = wp[0]; g1 = wp[1];
        }

        __syncthreads();   // dscnt wait + barrier: buffer p complete for everyone

        // A fragment: lanes 0-15 K {0..15}; lanes 16-31 K {16..31}
        const _Float16* arp = &LAp[(wid * 16 + m16) * 40];
        v8h fa0 = *(const v8h*)(arp + (hi ? 16 : 0));
        v8h fa1 = *(const v8h*)(arp + (hi ? 24 : 8));
        v16h af = __builtin_shufflevector(fa0, fa1, 0,1,2,3,4,5,6,7,8,9,10,11,12,13,14,15);
        const int kB = hi ? 16 : 0;
#pragma unroll
        for (int j = 0; j < 4; ++j) {
            const _Float16* brp = &LBp[(j * 16 + m16) * 40 + kB];
            v8h fb0 = *(const v8h*)brp;
            v8h fb1 = *(const v8h*)(brp + 8);
            v16h bf = __builtin_shufflevector(fb0, fb1, 0,1,2,3,4,5,6,7,8,9,10,11,12,13,14,15);
            acc[j] = __builtin_amdgcn_wmma_f32_16x16x32_f16(false, af, false, bf,
                                                            (short)0, acc[j], false, false);
        }
        __syncthreads();   // compute done before buffer p is restaged (iter ks+2)
    }

    // --------- epilogue: bounce C tile through LDS for coalesced b128 stores ------
    const int orow = tid >> 1;            // 0..127
    const int oseg = (tid & 1) << 4;      // 0 or 16
#pragma unroll
    for (int half = 0; half < 2; ++half) {
        __syncthreads();                  // LDS tiles no longer needed / prev half done
#pragma unroll
        for (int jj = 0; jj < 2; ++jj) {
            const int j = half * 2 + jj;
#pragma unroll
            for (int r = 0; r < 8; ++r)
                CT[(wid * 16 + r + hi * 8) * 33 + jj * 16 + m16] = acc[j][r];
        }
        __syncthreads();
        const float* src = &CT[orow * 33 + oseg];
        const int colbase = blockN + half * 32 + oseg;
        float out[16];
#pragma unroll
        for (int u = 0; u < 16; ++u)
            out[u] = act_apply<ACT>(src[u] + bias[colbase + u]);
        float4* dst = (float4*)(C + (size_t)(blockM + orow) * 512 + colbase);
#pragma unroll
        for (int q = 0; q < 4; ++q) {
            float4 o; o.x = out[q*4]; o.y = out[q*4+1]; o.z = out[q*4+2]; o.w = out[q*4+3];
            dst[q] = o;
        }
    }
}

// ----------------------------------------------------- linear attention (in-place Q)
__global__ __launch_bounds__(64) void attn_kernel(float* __restrict__ Q,
                                                  const float* __restrict__ K,
                                                  const float* __restrict__ V) {
    const int e = threadIdx.x;           // 0..63 (contiguous -> coalesced)
    const int b = blockIdx.x >> 3;
    const int h = blockIdx.x & 7;
    const size_t base = (size_t)b * 256 * 512 + h * 64 + e;
    float kv = 0.0f, ks = 0.0f;
    for (int s = 0; s < 256; ++s) {
        float kk = K[base + (size_t)s * 512];
        kv += kk * V[base + (size_t)s * 512];
        ks += kk;
    }
    for (int s = 0; s < 256; ++s) {
        float q = Q[base + (size_t)s * 512];
        Q[base + (size_t)s * 512] = q * kv / (q * ks + 1e-6f);
    }
}

// ---------------------------------------------------- out = LN(a + b) * s + bias
__global__ __launch_bounds__(256) void add_ln_kernel(const float* __restrict__ A,
                                                     const float* __restrict__ Bv,
                                                     const float* __restrict__ sc,
                                                     const float* __restrict__ bb,
                                                     float* __restrict__ out) {
    __shared__ float sh1[256], sh2[256];
    const int t = threadIdx.x;
    const size_t off = (size_t)blockIdx.x * 512;
    float x0 = A[off + t]       + Bv[off + t];
    float x1 = A[off + t + 256] + Bv[off + t + 256];
    sh1[t] = x0 + x1;
    sh2[t] = x0 * x0 + x1 * x1;
    __syncthreads();
    for (int st = 128; st > 0; st >>= 1) {
        if (t < st) { sh1[t] += sh1[t + st]; sh2[t] += sh2[t + st]; }
        __syncthreads();
    }
    float mean = sh1[0] * (1.0f / 512.0f);
    float var  = sh2[0] * (1.0f / 512.0f) - mean * mean;
    float inv  = rsqrtf(var + 1e-6f);
    out[off + t]       = (x0 - mean) * inv * sc[t]       + bb[t];
    out[off + t + 256] = (x1 - mean) * inv * sc[t + 256] + bb[t + 256];
}

// ------------------- VAE head: [16,131072] @ [131072,512], split-K=8, HBM-bound
__global__ __launch_bounds__(256) void final_wmma(const float* __restrict__ H,
                                                  const float* __restrict__ Wm,
                                                  const float* __restrict__ Wlv,
                                                  float* __restrict__ part) {
    __shared__ __align__(16) _Float16 LB[8][16 * 40];  // per-wave transposed [n][k]
    __shared__ float redbuf[8 * 256];
    const int tid = threadIdx.x, lane = tid & 31, wid = tid >> 5;
    const int bid = blockIdx.x;
    const int nt = bid & 31;
    const int outSel = (bid >> 5) & 1;
    const int kspl = bid >> 6;
    const float* __restrict__ W = outSel ? Wlv : Wm;
    const int n0  = nt * 16;
    const int kb0 = kspl * 16384 + wid * 2048;
    const int m16 = lane & 15, hi = lane >> 4;
    const int off0 = hi ? 16 : 0, off1 = hi ? 24 : 8;
    _Float16* lb = LB[wid];
    const float* apb = H + (size_t)m16 * SD;
    v8f acc = {};

    float4 g0, g1, g2, g3;           // weight-row regs (coalesced b128 stream)
    float4 a00, a01, a10, a11;       // A fragment regs
    {
        const float4* wp = (const float4*)(W + (size_t)(kb0 + lane) * 512 + n0);
        g0 = wp[0]; g1 = wp[1]; g2 = wp[2]; g3 = wp[3];
        const float* ap = apb + kb0;
        a00 = *(const float4*)(ap + off0); a01 = *(const float4*)(ap + off0 + 4);
        a10 = *(const float4*)(ap + off1); a11 = *(const float4*)(ap + off1 + 4);
    }

    for (int s = 0; s < 64; ++s) {
        // commit weight tile to per-wave LDS (same-wave DS ops are in-order)
        v8h h0 = cvt8(g0, g1), h1 = cvt8(g2, g3);
#pragma unroll
        for (int j = 0; j < 8; ++j) {
            lb[j * 40 + lane]       = h0[j];
            lb[(j + 8) * 40 + lane] = h1[j];
        }
        v16h af = __builtin_shufflevector(cvt8(a00, a01), cvt8(a10, a11),
                                          0,1,2,3,4,5,6,7,8,9,10,11,12,13,14,15);

        // issue next step's global loads before computing (latency under WMMA)
        if (s < 63) {
            const int kk = kb0 + (s + 1) * 32;
            const float4* wp = (const float4*)(W + (size_t)(kk + lane) * 512 + n0);
            g0 = wp[0]; g1 = wp[1]; g2 = wp[2]; g3 = wp[3];
            const float* ap = apb + kk;
            a00 = *(const float4*)(ap + off0); a01 = *(const float4*)(ap + off0 + 4);
            a10 = *(const float4*)(ap + off1); a11 = *(const float4*)(ap + off1 + 4);
            if (kk + 544 <= SD)
                __builtin_prefetch(W + (size_t)(kk + 512) * 512 + n0, 0, 1);
        }

        asm volatile("s_wait_dscnt 0x0" ::: "memory");  // wave-local LDS RAW
        const _Float16* brp = &lb[m16 * 40 + (hi ? 16 : 0)];
        v16h bf = __builtin_shufflevector(*(const v8h*)brp, *(const v8h*)(brp + 8),
                                          0,1,2,3,4,5,6,7,8,9,10,11,12,13,14,15);
        acc = __builtin_amdgcn_wmma_f32_16x16x32_f16(false, af, false, bf,
                                                     (short)0, acc, false, false);
    }
#pragma unroll
    for (int r = 0; r < 8; ++r) redbuf[wid * 256 + (r + hi * 8) * 16 + m16] = acc[r];
    __syncthreads();
    float sum = 0.0f;
#pragma unroll
    for (int ww = 0; ww < 8; ++ww) sum += redbuf[ww * 256 + tid];
    part[(size_t)bid * 256 + tid] = sum;
}

__global__ __launch_bounds__(256) void final_reduce(const float* __restrict__ part,
                                                    const float* __restrict__ bm,
                                                    const float* __restrict__ blv,
                                                    float* __restrict__ out) {
    const int gid = blockIdx.x * 256 + threadIdx.x;   // 0..16383
    const int outSel = gid >> 13;
    const int rem = gid & 8191;
    const int m = rem >> 9;
    const int n = rem & 511;
    const int nt = n >> 4, nn = n & 15;
    const int t = m * 16 + nn;
    float s = 0.0f;
#pragma unroll
    for (int k = 0; k < 8; ++k)
        s += part[(size_t)(k * 64 + outSel * 32 + nt) * 256 + t];
    s += (outSel ? blv : bm)[n];
    out[gid] = s;
}

// ---------------------------------------------------------------------------- host
extern "C" void kernel_launch(void* const* d_in, const int* in_sizes, int n_in,
                              void* d_out, int out_size, void* d_ws, size_t ws_size,
                              hipStream_t stream) {
    (void)in_sizes; (void)n_in; (void)out_size; (void)ws_size;
    const float* x       = (const float*)d_in[0];
    const float* pos_emb = (const float*)d_in[1];
    const float* W_in    = (const float*)d_in[2];
    const float* b_in    = (const float*)d_in[3];
    const float* Wq      = (const float*)d_in[4];
    const float* bq      = (const float*)d_in[5];
    const float* Wk      = (const float*)d_in[6];
    const float* bk      = (const float*)d_in[7];
    const float* Wv      = (const float*)d_in[8];
    const float* bv      = (const float*)d_in[9];
    const float* Wo      = (const float*)d_in[10];
    const float* bo      = (const float*)d_in[11];
    const float* ln1_s   = (const float*)d_in[12];
    const float* ln1_b   = (const float*)d_in[13];
    const float* W1      = (const float*)d_in[14];
    const float* b1      = (const float*)d_in[15];
    const float* W2      = (const float*)d_in[16];
    const float* b2      = (const float*)d_in[17];
    const float* ln2_s   = (const float*)d_in[18];
    const float* ln2_b   = (const float*)d_in[19];
    const float* Wm      = (const float*)d_in[20];
    const float* bm      = (const float*)d_in[21];
    const float* Wlv     = (const float*)d_in[22];
    const float* blv     = (const float*)d_in[23];

    float* ws = (float*)d_ws;
    float* Hb   = ws;                 // current activations
    float* Qb   = ws + 1 * NELEM;
    float* Kb   = ws + 2 * NELEM;
    float* Vb   = ws + 3 * NELEM;
    float* Tb   = ws + 4 * NELEM;     // layer input / residual
    float* part = ws + 5 * NELEM;     // split-K partials: 512 * 256 floats

    prep_kernel<<<NELEM / 1024, 256, 0, stream>>>(x, pos_emb, Hb);

    const dim3 gg(8, 32);  // N tiles x M tiles
    for (int i = 0; i < 2; ++i) {
        const size_t wOff = (size_t)i * 512 * 512;   // also H*D*DH for head weights
        const size_t bOff = (size_t)i * 512;
        // h = h @ W_in + b_in  (layer input / residual)
        gemm512_wmma<0, 0><<<gg, 256, 0, stream>>>(Hb, W_in + wOff, b_in + bOff, Tb);
        // per-head QKV projections
        gemm512_wmma<1, 2><<<gg, 256, 0, stream>>>(Tb, Wq + wOff, bq + bOff, Qb);
        gemm512_wmma<1, 2><<<gg, 256, 0, stream>>>(Tb, Wk + wOff, bk + bOff, Kb);
        gemm512_wmma<1, 0><<<gg, 256, 0, stream>>>(Tb, Wv + wOff, bv + bOff, Vb);
        // linear attention, att written into Qb
        attn_kernel<<<128, 64, 0, stream>>>(Qb, Kb, Vb);
        // h2 = gelu(att @ Wo + bo)
        gemm512_wmma<0, 1><<<gg, 256, 0, stream>>>(Qb, Wo + wOff, bo + bOff, Hb);
        // h = LN(h2 + res)
        add_ln_kernel<<<4096, 256, 0, stream>>>(Hb, Tb, ln1_s + bOff, ln1_b + bOff, Kb);
        // MLP
        gemm512_wmma<0, 1><<<gg, 256, 0, stream>>>(Kb, W1 + wOff, b1 + bOff, Vb);
        gemm512_wmma<0, 1><<<gg, 256, 0, stream>>>(Vb, W2 + wOff, b2 + bOff, Tb);
        add_ln_kernel<<<4096, 256, 0, stream>>>(Tb, Kb, ln2_s + bOff, ln2_b + bOff, Hb);
    }

    // VAE head: stream 536 MB of weights once (HBM-bound), deterministic split-K
    final_wmma<<<512, 256, 0, stream>>>(Hb, Wm, Wlv, part);
    final_reduce<<<64, 256, 0, stream>>>(part, bm, blv, (float*)d_out);
}
